// MPN_15444702396779
// MI455X (gfx1250) — compile-verified, matrix operand order checked by reading, not run on hardware
//
#include <hip/hip_runtime.h>
#include <hip/hip_bf16.h>

// ---------------------------------------------------------------------------
// Chemprop MPN on MI455X (gfx1250).
// bf16 messages + v_wmma_f32_16x16x32_bf16 GEMMs with fp32 accumulate.
// B operand staged through LDS (double-buffered, async-to-LDS when available);
// hidden padded 300 -> 320 so every stride is 16B aligned.
// ---------------------------------------------------------------------------

typedef __bf16 bf16;
typedef __attribute__((ext_vector_type(16))) __bf16 v16bf;
typedef __attribute__((ext_vector_type(8)))  __bf16 v8bf;
typedef __attribute__((ext_vector_type(8)))  float  v8f;
typedef __attribute__((ext_vector_type(2)))  int    v2i;

#define NUM_ATOMS 100000
#define NUM_BONDS 200000
#define MAX_NB    6
#define ATOM_FDIM 133
#define BOND_FDIM 147
#define HID       300
#define NUM_MOLS  8192

#define HP  320          // padded hidden (20 n-tiles of 16)
#define NT  20           // n-tiles total
#define NTW 10           // n-tiles per wave-strip (grid.y = 2)
#define KA_PAD 160       // padded atom fdim (5 k-steps of 32)
#define KB_PAD 160       // padded bond fdim (5 k-steps of 32)

#if defined(__gfx1250__) && __has_builtin(__builtin_amdgcn_global_load_async_to_lds_b64)
#define HAVE_ASYNC_COPY 1
#else
#define HAVE_ASYNC_COPY 0
#endif

#if HAVE_ASYNC_COPY
typedef v2i __attribute__((address_space(1)))* gptr_v2i;
typedef v2i __attribute__((address_space(3)))* lptr_v2i;

template <int N>
__device__ __forceinline__ void wait_asynccnt() {
#if __has_builtin(__builtin_amdgcn_s_wait_asynccnt)
  __builtin_amdgcn_s_wait_asynccnt(N);
#else
  asm volatile("s_wait_asynccnt %0" ::"i"(N) : "memory");
#endif
}
#endif

// --- A-operand fragment load (16-bit A 16x32 layout, ISA 7.12.2) -----------
// lane L: m = L&15 ; K pattern: base=(L>>4)*8 -> elems {base..+7, base+16..+23}
__device__ __forceinline__ v16bf load_a_frag(const bf16* __restrict__ Arow,
                                             int lda, int k0, int lane) {
  const int m  = lane & 15;
  const int kb = (lane >> 4) << 3;
  const bf16* p = Arow + (size_t)m * lda + k0 + kb;
  v8bf lo = *(const v8bf*)p;            // K = base .. base+7
  v8bf hi = *(const v8bf*)(p + 16);     // K = base+16 .. base+23
  v16bf r;
#pragma unroll
  for (int i = 0; i < 8; ++i) { r[i] = lo[i]; r[i + 8] = hi[i]; }
  return r;
}

// --- pack fp32 weight [K x N] row-major into B fragments -------------------
// fragment chunk (kt,nt): 32 lanes x 16 bf16, lane n = lane&15, same K pattern.
__global__ void pack_w(const float* __restrict__ W, bf16* __restrict__ out,
                       int K, int N, int ktiles) {
  int idx = blockIdx.x * 256 + threadIdx.x;
  int total = ktiles * NT * 512;
  if (idx >= total) return;
  int i    = idx & 15;
  int lane = (idx >> 4) & 31;
  int frag = idx >> 9;            // kt*NT + nt
  int nt   = frag % NT;
  int kt   = frag / NT;
  int kb   = (lane >> 4) << 3;
  int k    = kt * 32 + kb + (i < 8 ? i : i + 8);
  int n    = nt * 16 + (lane & 15);
  float v  = (k < K && n < N) ? W[(size_t)k * N + n] : 0.0f;
  out[idx] = (bf16)v;
}

// --- fp32 [rows x scols] -> bf16 [rows x dcols] zero-padded ----------------
__global__ void cvt_pad(const float* __restrict__ src, bf16* __restrict__ dst,
                        int rows, int scols, int dcols) {
  long long idx = (long long)blockIdx.x * 256 + threadIdx.x;
  if (idx >= (long long)rows * dcols) return;
  int r = (int)(idx / dcols);
  int c = (int)(idx % dcols);
  dst[idx] = (c < scols) ? (bf16)src[(size_t)r * scols + c] : (bf16)0.0f;
}

// --- WMMA GEMM: C[16-row tile x 160-col strip] per wave, B via LDS ---------
template <int HAS2, int HAS_ADDEND, int HAS_BIAS, int WRITE_F32, int WRITE_BF16,
          int ATOMIC_MEAN>
__global__ __launch_bounds__(256) void gemm_wmma(
    int Mtiles,
    const bf16* __restrict__ A1, int lda1, int ks1, const bf16* __restrict__ B1,
    const bf16* __restrict__ A2, int lda2, int ks2, const bf16* __restrict__ B2,
    const float* __restrict__ addend,   // fp32 [M x HP] (pre-activation residual)
    const float* __restrict__ bias,     // fp32 [HID]
    float* __restrict__ outF,           // fp32 [M x HP] (pre-activation)
    bf16*  __restrict__ outB,           // bf16 [M x HP] (relu)
    const int* __restrict__ mol_ids,
    float* __restrict__ outMean,        // fp32 [NUM_MOLS x HID]
    float* __restrict__ counts) {
  const int wave = threadIdx.x >> 5;
  const int lane = threadIdx.x & 31;
  int mt = blockIdx.x * 8 + wave;
  const bool active = (mt < Mtiles);
  if (mt >= Mtiles) mt = Mtiles - 1;        // clamp: keep wave for barriers/EXEC
  const int nt0  = blockIdx.y * NTW;
  const int row0 = mt * 16;

  // double-buffered B strip: one k-step = NTW contiguous 1KB fragments = 10KB
  __shared__ bf16 Bsh[2][NTW * 512];

  const int kstot = ks1 + (HAS2 ? ks2 : 0);

  auto bsrc = [&](int ks) -> const bf16* {
    if (!HAS2 || ks < ks1) return B1 + (size_t)(ks * NT + nt0) * 512;
    return B2 + (size_t)((ks - ks1) * NT + nt0) * 512;
  };
  auto loadA = [&](int ks) -> v16bf {
    if (!HAS2 || ks < ks1)
      return load_a_frag(A1 + (size_t)row0 * lda1, lda1, ks * 32, lane);
    return load_a_frag(A2 + (size_t)row0 * lda2, lda2, (ks - ks1) * 32, lane);
  };
  // cooperative copy of one 10KB strip: 1280 x 8B chunks, exactly 5 per thread
  auto stage = [&](int buf, const bf16* src) {
#if HAVE_ASYNC_COPY
#pragma unroll
    for (int it = 0; it < 5; ++it) {
      int c = threadIdx.x + it * 256;
      __builtin_amdgcn_global_load_async_to_lds_b64(
          (gptr_v2i)(src + c * 4), (lptr_v2i)(&Bsh[buf][c * 4]), 0, 0);
    }
#else
    const uint2* s = (const uint2*)src;
    uint2* d = (uint2*)&Bsh[buf][0];
#pragma unroll
    for (int it = 0; it < 5; ++it) {
      int c = threadIdx.x + it * 256;
      d[c] = s[c];
    }
#endif
  };

  v8f acc[NTW];
#pragma unroll
  for (int i = 0; i < NTW; ++i)
#pragma unroll
    for (int j = 0; j < 8; ++j) acc[i][j] = 0.0f;

  stage(0, bsrc(0));
  v16bf a = loadA(0);

  for (int ks = 0; ks < kstot; ++ks) {
    const int cur = ks & 1;
    if (ks + 1 < kstot) {
      stage(cur ^ 1, bsrc(ks + 1));
#if HAVE_ASYNC_COPY
      wait_asynccnt<5>();                 // current strip's 5 copies retired
#endif
    } else {
#if HAVE_ASYNC_COPY
      wait_asynccnt<0>();
#endif
    }
    __syncthreads();                      // strip[cur] visible to all waves
    v16bf an = (ks + 1 < kstot) ? loadA(ks + 1) : a;   // prefetch next A
#pragma unroll
    for (int nt = 0; nt < NTW; ++nt) {
      v16bf b = *(const v16bf*)(&Bsh[cur][nt * 512 + lane * 16]);
      acc[nt] = __builtin_amdgcn_wmma_f32_16x16x32_bf16(
          false, a, false, b, (short)0, acc[nt], false, false);
    }
    a = an;
    __syncthreads();                      // all waves done reading strip[cur]
  }

  if (!active) return;

  // Epilogue. C layout: lane -> n = lane&15, rows (lane>>4)*8 + v (ISA 7.12.2)
  const int nl = lane & 15;
  const int mb = (lane >> 4) << 3;
#pragma unroll
  for (int nt = 0; nt < NTW; ++nt) {
    const int col = (nt0 + nt) * 16 + nl;
#pragma unroll
    for (int v = 0; v < 8; ++v) {
      const int r = row0 + mb + v;
      float x = acc[nt][v];
      if (HAS_ADDEND) x += addend[(size_t)r * HP + col];
      if (HAS_BIAS && col < HID) x += bias[col];
      const float xr = x > 0.0f ? x : 0.0f;
      if (WRITE_F32)  outF[(size_t)r * HP + col] = x;        // pre-activation
      if (WRITE_BF16) outB[(size_t)r * HP + col] = (bf16)xr; // relu, bf16
      if (ATOMIC_MEAN) {
        const int mol = mol_ids[r];
        if (col < HID) atomicAdd(outMean + (size_t)mol * HID + col, xr);
        if (col == 0)  atomicAdd(counts + mol, 1.0f);
      }
    }
  }
}

// --- per-bond aggregation: agg[b] = sum_k msg[a2b[b2a[b]][k]] - msg[b2revb[b]]
// one thread per 8-column chunk (b128 loads/stores); 40 chunks per row
__global__ void gather_bonds(const bf16* __restrict__ msg,
                             const int* __restrict__ a2b,
                             const int* __restrict__ b2a,
                             const int* __restrict__ b2revb,
                             bf16* __restrict__ agg) {
  const int idx = blockIdx.x * 256 + threadIdx.x;   // NUM_BONDS*40 total, exact
  const int b = idx / 40;
  const int c = (idx % 40) * 8;
  const int a  = b2a[b];
  const int rb = b2revb[b];
  float acc[8];
  {
    const v8bf rv = *(const v8bf*)(msg + (size_t)rb * HP + c);
#pragma unroll
    for (int j = 0; j < 8; ++j) acc[j] = -(float)rv[j];
  }
#pragma unroll
  for (int k = 0; k < MAX_NB; ++k) {
    const int nb = a2b[a * MAX_NB + k];
    const v8bf v = *(const v8bf*)(msg + (size_t)nb * HP + c);
#pragma unroll
    for (int j = 0; j < 8; ++j) acc[j] += (float)v[j];
  }
  v8bf o;
#pragma unroll
  for (int j = 0; j < 8; ++j) o[j] = (bf16)acc[j];
  *(v8bf*)(agg + (size_t)b * HP + c) = o;
}

// --- per-atom aggregation: amsg[a] = sum_k msg[a2b[a][k]] ------------------
__global__ void gather_atoms(const bf16* __restrict__ msg,
                             const int* __restrict__ a2b,
                             bf16* __restrict__ amsg) {
  const int idx = blockIdx.x * 256 + threadIdx.x;   // NUM_ATOMS*40 total, exact
  const int a = idx / 40;
  const int c = (idx % 40) * 8;
  float acc[8];
#pragma unroll
  for (int j = 0; j < 8; ++j) acc[j] = 0.0f;
#pragma unroll
  for (int k = 0; k < MAX_NB; ++k) {
    const int nb = a2b[a * MAX_NB + k];
    const v8bf v = *(const v8bf*)(msg + (size_t)nb * HP + c);
#pragma unroll
    for (int j = 0; j < 8; ++j) acc[j] += (float)v[j];
  }
  v8bf o;
#pragma unroll
  for (int j = 0; j < 8; ++j) o[j] = (bf16)acc[j];
  *(v8bf*)(amsg + (size_t)a * HP + c) = o;
}

__global__ void zero_out(float* __restrict__ out, float* __restrict__ counts) {
  int idx = blockIdx.x * 256 + threadIdx.x;
  const int n1 = NUM_MOLS * HID;
  if (idx < n1) out[idx] = 0.0f;
  else if (idx < n1 + NUM_MOLS) counts[idx - n1] = 0.0f;
}

__global__ void div_mean(float* __restrict__ out, const float* __restrict__ counts) {
  int idx = blockIdx.x * 256 + threadIdx.x;
  if (idx >= NUM_MOLS * HID) return;
  out[idx] = out[idx] / fmaxf(counts[idx / HID], 1.0f);
}

// ---------------------------------------------------------------------------
extern "C" void kernel_launch(void* const* d_in, const int* in_sizes, int n_in,
                              void* d_out, int out_size, void* d_ws, size_t ws_size,
                              hipStream_t stream) {
  (void)in_sizes; (void)n_in; (void)out_size; (void)ws_size;
  const float* f_atoms = (const float*)d_in[0];
  const float* f_bonds = (const float*)d_in[1];
  const float* W_i     = (const float*)d_in[2];
  const float* W_h     = (const float*)d_in[3];
  const float* W_o     = (const float*)d_in[4];
  const float* b_o     = (const float*)d_in[5];
  const int*   a2b     = (const int*)d_in[6];
  const int*   b2a     = (const int*)d_in[7];
  const int*   b2revb  = (const int*)d_in[8];
  const int*   mol_ids = (const int*)d_in[9];
  float* out = (float*)d_out;

  char* ws = (char*)d_ws;
  size_t off = 0;
  auto carve = [&](size_t bytes) -> char* {
    char* p = ws + off;
    off = (off + bytes + 255) & ~(size_t)255;
    return p;
  };
  bf16* Wi_p   = (bf16*)carve((size_t)5  * NT * 512 * 2);
  bf16* Wh_p   = (bf16*)carve((size_t)10 * NT * 512 * 2);
  bf16* Woa_p  = (bf16*)carve((size_t)5  * NT * 512 * 2);
  bf16* Wom_p  = (bf16*)carve((size_t)10 * NT * 512 * 2);
  float* counts= (float*)carve((size_t)NUM_MOLS * 4);
  bf16* fb     = (bf16*)carve((size_t)NUM_BONDS * KB_PAD * 2);   // 64 MB
  bf16* fa     = (bf16*)carve((size_t)NUM_ATOMS * KA_PAD * 2);   // 32 MB
  float* inp   = (float*)carve((size_t)NUM_BONDS * HP * 4);      // 256 MB
  bf16* msg    = (bf16*)carve((size_t)NUM_BONDS * HP * 2);       // 128 MB
  bf16* agg    = (bf16*)carve((size_t)NUM_BONDS * HP * 2);       // 128 MB
  bf16* amsg   = (bf16*)carve((size_t)NUM_ATOMS * HP * 2);       // 64 MB

  // ---- weight packing (fragment order; zero-padded K/N) ----
  pack_w<<<(5 * NT * 512 + 255) / 256, 256, 0, stream>>>(W_i, Wi_p, BOND_FDIM, HID, 5);
  pack_w<<<(10 * NT * 512 + 255) / 256, 256, 0, stream>>>(W_h, Wh_p, HID, HID, 10);
  pack_w<<<(5 * NT * 512 + 255) / 256, 256, 0, stream>>>(W_o, Woa_p, ATOM_FDIM, HID, 5);
  pack_w<<<(10 * NT * 512 + 255) / 256, 256, 0, stream>>>(
      W_o + (size_t)ATOM_FDIM * HID, Wom_p, HID, HID, 10);

  // ---- feature conversion to padded bf16 ----
  {
    long long n = (long long)NUM_BONDS * KB_PAD;
    cvt_pad<<<(unsigned)((n + 255) / 256), 256, 0, stream>>>(f_bonds, fb, NUM_BONDS, BOND_FDIM, KB_PAD);
  }
  {
    long long n = (long long)NUM_ATOMS * KA_PAD;
    cvt_pad<<<(unsigned)((n + 255) / 256), 256, 0, stream>>>(f_atoms, fa, NUM_ATOMS, ATOM_FDIM, KA_PAD);
  }

  const int MtB = NUM_BONDS / 16;   // 12500
  const int MtA = NUM_ATOMS / 16;   // 6250
  dim3 gB((MtB + 7) / 8, 2), gA((MtA + 7) / 8, 2);

  // ---- inp = f_bonds @ W_i (fp32) ; msg = relu(inp) (bf16) ----
  gemm_wmma<0,0,0,1,1,0><<<gB, 256, 0, stream>>>(
      MtB, fb, KB_PAD, 5, Wi_p, nullptr, 0, 0, nullptr,
      nullptr, nullptr, inp, msg, nullptr, nullptr, nullptr);

  // ---- 3 message-passing iterations ----
  for (int d = 0; d < 3; ++d) {
    gather_bonds<<<NUM_BONDS * 40 / 256, 256, 0, stream>>>(msg, a2b, b2a, b2revb, agg);
    gemm_wmma<0,1,0,0,1,0><<<gB, 256, 0, stream>>>(
        MtB, agg, HP, 10, Wh_p, nullptr, 0, 0, nullptr,
        inp, nullptr, nullptr, msg, nullptr, nullptr, nullptr);
  }

  // ---- atom readout: relu([f_atoms, amsg] @ W_o + b_o), fused segment-sum ----
  gather_atoms<<<NUM_ATOMS * 40 / 256, 256, 0, stream>>>(msg, a2b, amsg);
  zero_out<<<(NUM_MOLS * HID + NUM_MOLS + 255) / 256, 256, 0, stream>>>(out, counts);
  gemm_wmma<1,0,1,0,0,1><<<gA, 256, 0, stream>>>(
      MtA, fa, KA_PAD, 5, Woa_p, amsg, HP, 10, Wom_p,
      nullptr, b_o, nullptr, nullptr, mol_ids, out, counts);
  div_mean<<<(NUM_MOLS * HID + 255) / 256, 256, 0, stream>>>(out, counts);
}
